// SelfAttentionBlock_10720238371215
// MI455X (gfx1250) — matrix operand order repaired and checked by reading, not compiled, not run on hardware
//
#include <hip/hip_runtime.h>
#include <hip/hip_bf16.h>

typedef __attribute__((ext_vector_type(16))) _Float16 v16h;
typedef __attribute__((ext_vector_type(8)))  _Float16 v8h;
typedef __attribute__((ext_vector_type(8)))  float    v8f;

#define B_   8
#define C_   512
#define N_   1024
#define NH   8
#define HD   64
#define NG   32
#define CPG  16        // C_/NG
#define EPS  1e-5f

// ---------------------------------------------------------------- helpers

__device__ __forceinline__ v8f wmma_f16(v16h a, v16h b, v8f c) {
    return __builtin_amdgcn_wmma_f32_16x16x32_f16(
        /*neg_a=*/false, a, /*neg_b=*/false, b,
        /*c_mod=*/(short)0, c, /*reuse_a=*/false, /*reuse_b=*/false);
}

// A-frag (16x32, f16) from row-major [M][K] matrix, leading dim ld.
// lane r=lane&15 holds row m0+r; elems 0..7 -> k0+s*8+e ; elems 8..15 -> k0+16+s*8+e
__device__ __forceinline__ v16h load_a_frag(const _Float16* A, int ld, int m0, int k0, int lane) {
    const int r = lane & 15, s = lane >> 4;
    const _Float16* p = A + (size_t)(m0 + r) * ld + k0 + s * 8;
    v8h lo = *(const v8h*)p;
    v8h hi = *(const v8h*)(p + 16);
    v16h a;
#pragma unroll
    for (int e = 0; e < 8; ++e) { a[e] = lo[e]; a[e + 8] = hi[e]; }
    return a;
}

// B-frag (32x16, f16) from memory stored column-per-row: BT[n][k] row-major, ld = K stride.
// lane holds column n = n0 + (lane&15); elems 0..7 -> k0+s*8+e ; elems 8..15 -> k0+16+s*8+e
__device__ __forceinline__ v16h load_b_frag(const _Float16* BT, int ld, int n0, int k0, int lane) {
    const int r = lane & 15, s = lane >> 4;
    const _Float16* p = BT + (size_t)(n0 + r) * ld + k0 + s * 8;
    v8h lo = *(const v8h*)p;
    v8h hi = *(const v8h*)(p + 16);
    v16h b;
#pragma unroll
    for (int e = 0; e < 8; ++e) { b[e] = lo[e]; b[e + 8] = hi[e]; }
    return b;
}

// ---------------------------------------------------------------- fp32 -> fp16

__global__ __launch_bounds__(256) void cvt_f16_kernel(const float* __restrict__ src,
                                                      _Float16* __restrict__ dst, int n) {
    int i = blockIdx.x * 256 + threadIdx.x;
    if (i < n) dst[i] = (_Float16)src[i];
}

// ---------------------------------------------------------------- GroupNorm -> hT[b][n][c] (f16)

__global__ __launch_bounds__(256) void gn_kernel(const float* __restrict__ x,
                                                 const float* __restrict__ w,
                                                 const float* __restrict__ bgn,
                                                 _Float16* __restrict__ hT) {
    const int bg = blockIdx.x;
    const int b = bg >> 5, g = bg & 31;
    const int tid = threadIdx.x;
    const float* src = x + ((size_t)b * C_ + g * CPG) * N_;
    __shared__ float s1[256], s2[256];
    float a = 0.f, q = 0.f;
    for (int i = tid; i < CPG * N_; i += 256) { float v = src[i]; a += v; q += v * v; }
    s1[tid] = a; s2[tid] = q;
    __syncthreads();
    for (int off = 128; off > 0; off >>= 1) {
        if (tid < off) { s1[tid] += s1[tid + off]; s2[tid] += s2[tid + off]; }
        __syncthreads();
    }
    const float inv_n = 1.0f / (float)(CPG * N_);
    const float mean = s1[0] * inv_n;
    const float var  = s2[0] * inv_n - mean * mean;
    const float rstd = rsqrtf(var + EPS);
    for (int i = tid; i < CPG * N_; i += 256) {
        const int cl = i >> 10, n = i & (N_ - 1);
        const int c = g * CPG + cl;
        const float v = (src[i] - mean) * rstd * w[c] + bgn[c];
        hT[((size_t)b * N_ + n) * C_ + c] = (_Float16)v;
    }
}

// ---------------------------------------------------------------- QKV GEMM
// D(1536x1024) = Wqkv(1536x512) x h(512x1024); epilogue scatters q (x 1/8), k, v.

__global__ __launch_bounds__(256) void gemm_qkv_kernel(const _Float16* __restrict__ Wh,
                                                       const _Float16* __restrict__ hT,
                                                       const float* __restrict__ qkv_b,
                                                       _Float16* __restrict__ qT,
                                                       _Float16* __restrict__ kT,
                                                       _Float16* __restrict__ vN) {
    const int lane = threadIdx.x, wave = threadIdx.y;
    const int b  = blockIdx.z;
    const int n0 = blockIdx.x * 32;
    const int m0 = blockIdx.y * 256 + wave * 32;
    const _Float16* Bp = hT + (size_t)b * N_ * C_;

    v8f acc[2][2] = {};
    for (int k0 = 0; k0 < C_; k0 += 32) {
        v16h a0 = load_a_frag(Wh, C_, m0,      k0, lane);
        v16h a1 = load_a_frag(Wh, C_, m0 + 16, k0, lane);
        v16h b0 = load_b_frag(Bp, C_, n0,      k0, lane);
        v16h b1 = load_b_frag(Bp, C_, n0 + 16, k0, lane);
        acc[0][0] = wmma_f16(a0, b0, acc[0][0]);
        acc[0][1] = wmma_f16(a0, b1, acc[0][1]);
        acc[1][0] = wmma_f16(a1, b0, acc[1][0]);
        acc[1][1] = wmma_f16(a1, b1, acc[1][1]);
    }

    const int r = lane & 15, s = lane >> 4;
#pragma unroll
    for (int tm = 0; tm < 2; ++tm)
#pragma unroll
        for (int tn = 0; tn < 2; ++tn)
#pragma unroll
            for (int v = 0; v < 8; ++v) {
                const int m = m0 + tm * 16 + v + 8 * s;
                const int n = n0 + tn * 16 + r;
                const float val = acc[tm][tn][v] + qkv_b[m];
                const int sel = m >> 9;              // 0=q 1=k 2=v
                const int rr  = m & 511;
                const int head = rr >> 6, d = rr & 63;
                const size_t bh = (size_t)(b * NH + head);
                if (sel == 0)      qT[(bh * N_ + n) * HD + d] = (_Float16)(val * 0.125f);
                else if (sel == 1) kT[(bh * N_ + n) * HD + d] = (_Float16)val;
                else               vN[(bh * HD + d) * N_ + n] = (_Float16)val;
            }
}

// ---------------------------------------------------------------- flash attention
// per wave: 32 query rows, online softmax over j in blocks of 32, out = P*V via LDS re-layout.

__global__ __launch_bounds__(256) void attn_kernel(const _Float16* __restrict__ qT,
                                                   const _Float16* __restrict__ kT,
                                                   const _Float16* __restrict__ vN,
                                                   _Float16* __restrict__ oT) {
    __shared__ _Float16 ldsP[8][32][32];
    const int lane = threadIdx.x, wave = threadIdx.y;
    const int bh = blockIdx.y;
    const int b = bh >> 3, h = bh & 7;
    const int i0 = (blockIdx.x * 8 + wave) * 32;
    const _Float16* q = qT + (size_t)bh * N_ * HD;
    const _Float16* k = kT + (size_t)bh * N_ * HD;
    const _Float16* v = vN + (size_t)bh * HD * N_;
    const int r = lane & 15, s = lane >> 4;

    v16h aq[2][2];
#pragma unroll
    for (int it = 0; it < 2; ++it)
#pragma unroll
        for (int c = 0; c < 2; ++c)
            aq[it][c] = load_a_frag(q, HD, i0 + it * 16, c * 32, lane);

    v8f acco[2][4] = {};
    float mrun[2][8], lsum[2][8];
#pragma unroll
    for (int it = 0; it < 2; ++it)
#pragma unroll
        for (int e = 0; e < 8; ++e) { mrun[it][e] = -1e30f; lsum[it][e] = 0.f; }

    for (int j0 = 0; j0 < N_; j0 += 32) {
        v16h kb[2][2];
#pragma unroll
        for (int jt = 0; jt < 2; ++jt)
#pragma unroll
            for (int c = 0; c < 2; ++c)
                kb[jt][c] = load_b_frag(k, HD, j0 + jt * 16, c * 32, lane);

        v8f S[2][2] = {};
#pragma unroll
        for (int it = 0; it < 2; ++it)
#pragma unroll
            for (int jt = 0; jt < 2; ++jt) {
                S[it][jt] = wmma_f16(aq[it][0], kb[jt][0], S[it][jt]);
                S[it][jt] = wmma_f16(aq[it][1], kb[jt][1], S[it][jt]);
            }

#pragma unroll
        for (int it = 0; it < 2; ++it) {
            float mv[8];
#pragma unroll
            for (int e = 0; e < 8; ++e) mv[e] = fmaxf(S[it][0][e], S[it][1][e]);
#pragma unroll
            for (int off = 1; off < 16; off <<= 1)
#pragma unroll
                for (int e = 0; e < 8; ++e)
                    mv[e] = fmaxf(mv[e], __shfl_xor(mv[e], off, 32));

            float corr[8], p0[8], p1[8], rs[8];
#pragma unroll
            for (int e = 0; e < 8; ++e) {
                const float mn = fmaxf(mrun[it][e], mv[e]);
                corr[e] = __expf(mrun[it][e] - mn);
                mrun[it][e] = mn;
                p0[e] = __expf(S[it][0][e] - mn);
                p1[e] = __expf(S[it][1][e] - mn);
                rs[e] = p0[e] + p1[e];
            }
#pragma unroll
            for (int off = 1; off < 16; off <<= 1)
#pragma unroll
                for (int e = 0; e < 8; ++e)
                    rs[e] += __shfl_xor(rs[e], off, 32);
#pragma unroll
            for (int e = 0; e < 8; ++e)
                lsum[it][e] = lsum[it][e] * corr[e] + rs[e];
#pragma unroll
            for (int dt = 0; dt < 4; ++dt)
#pragma unroll
                for (int e = 0; e < 8; ++e)
                    acco[it][dt][e] *= corr[e];
            // spill P tile to per-wave LDS in [row][k] layout
#pragma unroll
            for (int e = 0; e < 8; ++e) {
                const int row = it * 16 + e + 8 * s;
                ldsP[wave][row][r]      = (_Float16)p0[e];
                ldsP[wave][row][16 + r] = (_Float16)p1[e];
            }
        }
        asm volatile("s_wait_dscnt 0" ::: "memory");   // same-wave LDS store->load ordering

        v16h pa[2];
#pragma unroll
        for (int it = 0; it < 2; ++it) {
            const _Float16* pp = &ldsP[wave][it * 16 + r][0];
            v8h lo = *(const v8h*)(pp + s * 8);
            v8h hi = *(const v8h*)(pp + 16 + s * 8);
#pragma unroll
            for (int e = 0; e < 8; ++e) { pa[it][e] = lo[e]; pa[it][e + 8] = hi[e]; }
        }
#pragma unroll
        for (int dt = 0; dt < 4; ++dt) {
            v16h vb = load_b_frag(v, N_, dt * 16, j0, lane);   // B[k=j][n=d] from v[d][j]
            acco[0][dt] = wmma_f16(pa[0], vb, acco[0][dt]);
            acco[1][dt] = wmma_f16(pa[1], vb, acco[1][dt]);
        }
    }

#pragma unroll
    for (int it = 0; it < 2; ++it)
#pragma unroll
        for (int dt = 0; dt < 4; ++dt)
#pragma unroll
            for (int e = 0; e < 8; ++e) {
                const float val = acco[it][dt][e] / lsum[it][e];
                const int i = i0 + it * 16 + e + 8 * s;
                const int d = dt * 16 + r;
                oT[((size_t)b * N_ + i) * C_ + h * HD + d] = (_Float16)val;
            }
}

// ---------------------------------------------------------------- proj GEMM + bias + residual

__global__ __launch_bounds__(256) void gemm_proj_kernel(const _Float16* __restrict__ Wh,
                                                        const _Float16* __restrict__ oT,
                                                        const float* __restrict__ proj_b,
                                                        const float* __restrict__ x,
                                                        float* __restrict__ out) {
    const int lane = threadIdx.x, wave = threadIdx.y;
    const int b  = blockIdx.z;
    const int n0 = blockIdx.x * 32;
    const int m0 = blockIdx.y * 256 + wave * 32;
    const _Float16* Bp = oT + (size_t)b * N_ * C_;

    v8f acc[2][2] = {};
    for (int k0 = 0; k0 < C_; k0 += 32) {
        v16h a0 = load_a_frag(Wh, C_, m0,      k0, lane);
        v16h a1 = load_a_frag(Wh, C_, m0 + 16, k0, lane);
        v16h b0 = load_b_frag(Bp, C_, n0,      k0, lane);
        v16h b1 = load_b_frag(Bp, C_, n0 + 16, k0, lane);
        acc[0][0] = wmma_f16(a0, b0, acc[0][0]);
        acc[0][1] = wmma_f16(a0, b1, acc[0][1]);
        acc[1][0] = wmma_f16(a1, b0, acc[1][0]);
        acc[1][1] = wmma_f16(a1, b1, acc[1][1]);
    }

    const int r = lane & 15, s = lane >> 4;
#pragma unroll
    for (int tm = 0; tm < 2; ++tm)
#pragma unroll
        for (int tn = 0; tn < 2; ++tn)
#pragma unroll
            for (int v = 0; v < 8; ++v) {
                const int m = m0 + tm * 16 + v + 8 * s;
                const int n = n0 + tn * 16 + r;
                const size_t idx = ((size_t)b * C_ + m) * N_ + n;
                out[idx] = x[idx] + acc[tm][tn][v] + proj_b[m];
            }
}

// ---------------------------------------------------------------- launcher

extern "C" void kernel_launch(void* const* d_in, const int* in_sizes, int n_in,
                              void* d_out, int out_size, void* d_ws, size_t ws_size,
                              hipStream_t stream) {
    (void)in_sizes; (void)n_in; (void)out_size; (void)ws_size;
    const float* x      = (const float*)d_in[0];
    const float* gn_w   = (const float*)d_in[1];
    const float* gn_b   = (const float*)d_in[2];
    const float* qkv_w  = (const float*)d_in[3];
    const float* qkv_b  = (const float*)d_in[4];
    const float* proj_w = (const float*)d_in[5];
    const float* proj_b = (const float*)d_in[6];
    float* out = (float*)d_out;

    char* ws = (char*)d_ws;
    const size_t OFF_WQKV  = 0;                         // 1536*512 f16
    const size_t OFF_WPROJ = OFF_WQKV  + (size_t)3 * C_ * C_ * 2;
    const size_t OFF_HT    = OFF_WPROJ + (size_t)C_ * C_ * 2;
    const size_t OFF_Q     = OFF_HT    + (size_t)B_ * N_ * C_ * 2;
    const size_t OFF_K     = OFF_Q     + (size_t)B_ * N_ * C_ * 2;
    const size_t OFF_V     = OFF_K     + (size_t)B_ * N_ * C_ * 2;
    const size_t OFF_OT    = OFF_V     + (size_t)B_ * N_ * C_ * 2;

    _Float16* wqkv_h = (_Float16*)(ws + OFF_WQKV);
    _Float16* wproj_h = (_Float16*)(ws + OFF_WPROJ);
    _Float16* hT = (_Float16*)(ws + OFF_HT);
    _Float16* qT = (_Float16*)(ws + OFF_Q);
    _Float16* kT = (_Float16*)(ws + OFF_K);
    _Float16* vN = (_Float16*)(ws + OFF_V);
    _Float16* oT = (_Float16*)(ws + OFF_OT);

    const int nWq = 3 * C_ * C_, nWp = C_ * C_;
    cvt_f16_kernel<<<(nWq + 255) / 256, 256, 0, stream>>>(qkv_w, wqkv_h, nWq);
    cvt_f16_kernel<<<(nWp + 255) / 256, 256, 0, stream>>>(proj_w, wproj_h, nWp);

    gn_kernel<<<B_ * NG, 256, 0, stream>>>(x, gn_w, gn_b, hT);

    gemm_qkv_kernel<<<dim3(N_ / 32, (3 * C_) / 256, B_), dim3(32, 8), 0, stream>>>(
        wqkv_h, hT, qkv_b, qT, kT, vN);

    attn_kernel<<<dim3(N_ / 256, B_ * NH), dim3(32, 8), 0, stream>>>(qT, kT, vN, oT);

    gemm_proj_kernel<<<dim3(N_ / 32, C_ / 256, B_), dim3(32, 8), 0, stream>>>(
        wproj_h, oT, proj_b, x, out);
}
